// BiMambaFFN_52956946760380
// MI455X (gfx1250) — compile-verified, matrix operand order checked by reading, not run on hardware
//
#include <hip/hip_runtime.h>
#include <hip/hip_bf16.h>

// ---------------- constants (match reference) ----------------
#define BB     4
#define SS     2048
#define DM     128      // D_MODEL
#define DI     256      // D_INNER
#define DSTATE 256      // D_STATE
#define MTOT   (BB*SS)  // 8192 rows for all GEMMs

typedef __attribute__((ext_vector_type(16))) __bf16 v16bf;
typedef __attribute__((ext_vector_type(8)))  __bf16 v8bf;
typedef __attribute__((ext_vector_type(8)))  float  v8f;

// ---------------------------------------------------------------------------
// bf16 WMMA fragment load, row-major source, per ISA 7.12.2 16-bit A layout:
// lane (l&15) = row, half = l>>4; per-lane data = elems [k0+8h .. +7] and
// [k0+8h+16 .. +23]  -> two global_load_b128.
// ---------------------------------------------------------------------------
__device__ __forceinline__ v16bf frag_load(const __bf16* p) {
  v8bf lo = *(const v8bf*)(p);
  v8bf hi = *(const v8bf*)(p + 16);
  v16bf v;
#pragma unroll
  for (int i = 0; i < 8; ++i) { v[i] = lo[i]; v[8 + i] = hi[i]; }
  return v;
}

// ---------------------------------------------------------------------------
// Generic GEMM: C[M,N] = A[M,K] @ W[Npad,K]^T (+bias).
// One wave computes a 16(M) x 64(N) strip: 4 f32 accumulators, A fragment
// reused 4x per K-chunk.  W buffer rows are zero-padded to a multiple of 64
// so the load path needs no per-lane guard (WMMA requires EXEC==all-ones);
// only stores are guarded by n < N.
// arev!=0 remaps A rows by time-flip within each batch of S=2048.
// ---------------------------------------------------------------------------
__global__ __launch_bounds__(256)
void wmma_gemm_bf16(const __bf16* __restrict__ A, int lda, int arev,
                    const __bf16* __restrict__ W, int ldw,
                    const float* __restrict__ bias,
                    float* __restrict__ C, int ldc,
                    int M, int N, int K)
{
  const int strips_n = (N + 63) >> 6;
  const int job = blockIdx.x * 8 + (threadIdx.x >> 5);
  const int tm = job / strips_n;
  const int sn = job - tm * strips_n;
  if (tm * 16 >= M) return;

  const int lane = threadIdx.x & 31;
  const int half = lane >> 4;

  int am = tm * 16 + (lane & 15);
  if (arev) { int b = am >> 11, s = am & (SS - 1); am = (b << 11) + (SS - 1 - s); }
  const __bf16* ap = A + (size_t)am * lda + 8 * half;

  const int n0 = sn * 64 + (lane & 15);           // rows n0 .. n0+48 exist (padded)
  const __bf16* wp = W + (size_t)n0 * ldw + 8 * half;
  const size_t wstep = (size_t)16 * ldw;

  v8f acc0, acc1, acc2, acc3;
#pragma unroll
  for (int i = 0; i < 8; ++i) { acc0[i] = 0.f; acc1[i] = 0.f; acc2[i] = 0.f; acc3[i] = 0.f; }

  for (int k0 = 0; k0 < K; k0 += 32) {
    __builtin_prefetch((const void*)(ap + k0 + 32), 0, 1);   // global_prefetch_b8
    __builtin_prefetch((const void*)(wp + k0 + 32), 0, 1);
    v16bf a  = frag_load(ap + k0);
    v16bf b0 = frag_load(wp + k0);
    v16bf b1 = frag_load(wp + wstep     + k0);
    v16bf b2 = frag_load(wp + wstep * 2 + k0);
    v16bf b3 = frag_load(wp + wstep * 3 + k0);
    acc0 = __builtin_amdgcn_wmma_f32_16x16x32_bf16(false, a, false, b0, (short)0, acc0, false, false);
    acc1 = __builtin_amdgcn_wmma_f32_16x16x32_bf16(false, a, false, b1, (short)0, acc1, false, false);
    acc2 = __builtin_amdgcn_wmma_f32_16x16x32_bf16(false, a, false, b2, (short)0, acc2, false, false);
    acc3 = __builtin_amdgcn_wmma_f32_16x16x32_bf16(false, a, false, b3, (short)0, acc3, false, false);
  }

  float* cp = C + (size_t)(tm * 16 + 8 * half) * ldc;
#pragma unroll
  for (int t = 0; t < 4; ++t) {
    const v8f& acc = (t == 0) ? acc0 : (t == 1) ? acc1 : (t == 2) ? acc2 : acc3;
    const int n = sn * 64 + t * 16 + (lane & 15);
    if (n < N) {
      const float bv = bias ? bias[n] : 0.f;
#pragma unroll
      for (int r = 0; r < 8; ++r) cp[(size_t)r * ldc + n] = acc[r] + bv;
    }
  }
}

// ---------------------------------------------------------------------------
// f32 -> bf16 conversion with trailing zero-pad (npad_total >= n_total).
// Pad region is whole trailing rows, so flat-index comparison suffices.
// ---------------------------------------------------------------------------
__global__ void cvt_bf16_kernel(const float* __restrict__ s, __bf16* __restrict__ d,
                                int n_total, int npad_total) {
  int i = blockIdx.x * blockDim.x + threadIdx.x;
  if (i < npad_total) d[i] = (i < n_total) ? (__bf16)s[i] : (__bf16)0.f;
}

// ---------------------------------------------------------------------------
// xi = silu(causal_dwconv4(xi_raw)); write f32 (scan u) + bf16 (next GEMM A)
// xi_raw lives in xz buffer (row stride ldx=512, cols 0..255)
// ---------------------------------------------------------------------------
__global__ void conv_silu_kernel(const float* __restrict__ xi, int ldx,
                                 const float* __restrict__ w,
                                 const float* __restrict__ cb,
                                 float* __restrict__ uo, __bf16* __restrict__ ub)
{
  int idx = blockIdx.x * blockDim.x + threadIdx.x;   // m*DI + c
  if (idx >= MTOT * DI) return;
  int c = idx & (DI - 1);
  int m = idx >> 8;
  int s = m & (SS - 1);
  const float* base = xi + (size_t)m * ldx + c;
  float acc = cb[c];
#pragma unroll
  for (int k = 0; k < 4; ++k) {
    int sk = s + k - 3;
    if (sk >= 0) acc += base[(long)(k - 3) * ldx] * w[c * 4 + k];
  }
  float sg = 1.f / (1.f + __expf(-acc));
  float v = acc * sg;
  uo[(size_t)m * DI + c] = v;
  ub[(size_t)m * DI + c] = (__bf16)v;
}

// ---------------------------------------------------------------------------
// dt = softplus(dt_raw @ Wdt^T + bdt); dt_raw = xdbc[:, :8]
// ---------------------------------------------------------------------------
__global__ void dt_kernel(const float* __restrict__ xdbc, int ldx,
                          const float* __restrict__ Wdt,
                          const float* __restrict__ bdt,
                          float* __restrict__ dt)
{
  int idx = blockIdx.x * blockDim.x + threadIdx.x;   // m*DI + c
  if (idx >= MTOT * DI) return;
  int c = idx & (DI - 1);
  int m = idx >> 8;
  const float* r = xdbc + (size_t)m * ldx;
  float acc = bdt[c];
#pragma unroll
  for (int k = 0; k < 8; ++k) acc += r[k] * Wdt[c * 8 + k];
  dt[idx] = (acc > 20.f) ? acc : log1pf(__expf(acc));
}

// ---------------------------------------------------------------------------
// Selective scan, fused gate.  Block = 256 thr = 8 waves = 8 channels of one
// batch.  Wave lane j owns states n = j + 32q, q=0..7.  B_t/C_t rows staged
// in LDS in tiles of 8 timesteps (shared by the 8 channel-waves).
// Epilogue: y = scan + u*D;  yg = y * silu(z)  -> bf16 for Wout GEMM.
// ---------------------------------------------------------------------------
__global__ __launch_bounds__(256)
void scan_kernel(const float* __restrict__ dt,
                 const float* __restrict__ u,
                 const float* __restrict__ Bc, int ldb,
                 const float* __restrict__ Cc, int ldcc,
                 const float* __restrict__ z,  int ldz,
                 const float* __restrict__ Alog,
                 const float* __restrict__ Dp,
                 __bf16* __restrict__ yg)
{
  __shared__ float sB[8 * DSTATE];
  __shared__ float sC[8 * DSTATE];
  const int wave = threadIdx.x >> 5;
  const int lane = threadIdx.x & 31;
  const int b = blockIdx.x >> 5;                  // 32 blocks per batch
  const int d = ((blockIdx.x & 31) << 3) + wave;  // channel

  float a[8], h[8];
#pragma unroll
  for (int q = 0; q < 8; ++q) {
    a[q] = -__expf(Alog[(size_t)d * DSTATE + lane + 32 * q]);
    h[q] = 0.f;
  }
  const float Dv = Dp[d];

  const float* dtb = dt + (size_t)b * SS * DI + d;
  const float* ub  = u  + (size_t)b * SS * DI + d;
  const float* Bb  = Bc + (size_t)b * SS * ldb;
  const float* Cb  = Cc + (size_t)b * SS * ldcc;
  const float* zb  = z  + (size_t)b * SS * ldz + d;
  __bf16* yb = yg + (size_t)b * SS * DI + d;

  for (int t0 = 0; t0 < SS; t0 += 8) {
    __syncthreads();
#pragma unroll
    for (int i = 0; i < 8; ++i) {
      sB[i * DSTATE + threadIdx.x] = Bb[(size_t)(t0 + i) * ldb  + threadIdx.x];
      sC[i * DSTATE + threadIdx.x] = Cb[(size_t)(t0 + i) * ldcc + threadIdx.x];
    }
    __syncthreads();
#pragma unroll 1
    for (int tt = 0; tt < 8; ++tt) {
      const int t = t0 + tt;
      const float dtv = dtb[(size_t)t * DI];
      const float uv  = ub [(size_t)t * DI];
      const float du  = dtv * uv;
      float acc = 0.f;
#pragma unroll
      for (int q = 0; q < 8; ++q) {
        const int n = lane + 32 * q;
        h[q] = h[q] * __expf(dtv * a[q]) + du * sB[tt * DSTATE + n];
        acc += h[q] * sC[tt * DSTATE + n];
      }
#pragma unroll
      for (int o = 16; o > 0; o >>= 1) acc += __shfl_xor(acc, o, 32);
      if (lane == 0) {
        const float yv = acc + uv * Dv;
        const float zv = zb[(size_t)t * ldz];
        const float sg = 1.f / (1.f + __expf(-zv));
        yb[(size_t)t * DI] = (__bf16)(yv * (zv * sg));
      }
    }
  }
}

// ---------------------------------------------------------------------------
// residual + per-channel scale, writes bf16 half of the concat buffer.
// hcat[:, 0:128] = xf, hcat[:, 128:256] = xb (xb stays time-flipped, as ref).
// ---------------------------------------------------------------------------
__global__ void resid_kernel(const float* __restrict__ ytmp,
                             const float* __restrict__ x,
                             const float* __restrict__ scale,
                             int rev, int colbase,
                             __bf16* __restrict__ hcat)
{
  int idx = blockIdx.x * blockDim.x + threadIdx.x;   // m*DM + j
  if (idx >= MTOT * DM) return;
  int j = idx & (DM - 1);
  int m = idx >> 7;
  int b = m >> 11;
  int s = m & (SS - 1);
  int ts = rev ? (SS - 1 - s) : s;
  float v = x[((size_t)(b << 11) + ts) * DM + j] + ytmp[idx] * scale[j];
  hcat[(size_t)m * (2 * DM) + colbase + j] = (__bf16)v;
}

// ---------------------------------------------------------------------------
// 3-tap "same" dwconv over time on h1 (512 ch), then GLU: silu(x1)*x2 -> bf16
// ---------------------------------------------------------------------------
__global__ void dwglu_kernel(const float* __restrict__ h1,
                             const float* __restrict__ w,
                             const float* __restrict__ bb,
                             __bf16* __restrict__ glu)
{
  int idx = blockIdx.x * blockDim.x + threadIdx.x;   // m*256 + c
  if (idx >= MTOT * 256) return;
  int c = idx & 255;
  int m = idx >> 8;
  int s = m & (SS - 1);
  float v[2];
#pragma unroll
  for (int p = 0; p < 2; ++p) {
    int col = c + p * 256;
    float acc = bb[col];
#pragma unroll
    for (int k = 0; k < 3; ++k) {
      int sk = s + k - 1;
      if (sk >= 0 && sk < SS)
        acc += h1[((size_t)m + (k - 1)) * 512 + col] * w[col * 3 + k];
    }
    v[p] = acc;
  }
  float sg = 1.f / (1.f + __expf(-v[0]));
  glu[idx] = (__bf16)(v[0] * sg * v[1]);
}

// ---------------------------------------------------------------------------
// grouped RMS norm: 4 groups of 32 channels; group == one wave32.
// Block = 128 threads = one row, 4 waves, one group per wave.
// ---------------------------------------------------------------------------
__global__ __launch_bounds__(128)
void rms_kernel(const float* __restrict__ h2,
                const float* __restrict__ gamma,
                float* __restrict__ out)
{
  const int m = blockIdx.x;
  const int c = threadIdx.x;
  float v = h2[(size_t)m * DM + c];
  float ss = v * v;
#pragma unroll
  for (int o = 16; o > 0; o >>= 1) ss += __shfl_xor(ss, o, 32);
  float rms = __fsqrt_rn(ss) * 0.17677669529663687f;  // * 1/sqrt(32)
  out[(size_t)m * DM + c] = v / (rms + 1e-5f) * gamma[c];
}

// ---------------------------------------------------------------------------
// host orchestration
// ---------------------------------------------------------------------------
static inline int ceil_div(int a, int b) { return (a + b - 1) / b; }

static void launch_gemm(const __bf16* A, int lda, int arev,
                        const __bf16* W, int ldw, const float* bias,
                        float* C, int ldc, int M, int N, int K,
                        hipStream_t stream)
{
  int jobs = (M / 16) * ceil_div(N, 64);
  int blocks = ceil_div(jobs, 8);
  wmma_gemm_bf16<<<blocks, 256, 0, stream>>>(A, lda, arev, W, ldw, bias, C, ldc, M, N, K);
}

static void launch_cvt(const float* s, __bf16* d, int n, int npad, hipStream_t stream) {
  cvt_bf16_kernel<<<ceil_div(npad, 256), 256, 0, stream>>>(s, d, n, npad);
}

extern "C" void kernel_launch(void* const* d_in, const int* in_sizes, int n_in,
                              void* d_out, int out_size, void* d_ws, size_t ws_size,
                              hipStream_t stream)
{
  (void)in_sizes; (void)n_in; (void)out_size; (void)ws_size;

  const float* x = (const float*)d_in[0];
  const float* fscale  = (const float*)d_in[19];
  const float* bscale  = (const float*)d_in[20];
  const float* convf_w = (const float*)d_in[21];
  const float* convf_b = (const float*)d_in[22];
  const float* dw_w    = (const float*)d_in[23];
  const float* dw_b    = (const float*)d_in[24];
  const float* convo_w = (const float*)d_in[25];
  const float* convo_b = (const float*)d_in[26];
  const float* gamma   = (const float*)d_in[27];

  // ---- workspace carve-up (256B aligned slices) ----
  char* ws = (char*)d_ws;
  size_t off = 0;
  auto take = [&](size_t bytes) -> void* {
    void* p = ws + off;
    off += (bytes + 255) & ~(size_t)255;
    return p;
  };
  float*  xz    = (float*)take((size_t)MTOT * 512 * 4);  // xz; reused as h1
  float*  uf    = (float*)take((size_t)MTOT * 256 * 4);
  float*  dtv   = (float*)take((size_t)MTOT * 256 * 4);  // dt; reused as h2
  float*  xdbc  = (float*)take((size_t)MTOT * 528 * 4);  // N=520 padded to 528
  float*  ytmp  = (float*)take((size_t)MTOT * 128 * 4);
  __bf16* xbf    = (__bf16*)take((size_t)MTOT * 128 * 2);
  __bf16* ubf    = (__bf16*)take((size_t)MTOT * 256 * 2);
  __bf16* ygbf   = (__bf16*)take((size_t)MTOT * 256 * 2);
  __bf16* hcat   = (__bf16*)take((size_t)MTOT * 256 * 2);
  __bf16* glubf  = (__bf16*)take((size_t)MTOT * 256 * 2);
  __bf16* Winbf  = (__bf16*)take((size_t)512 * 128 * 2);
  __bf16* Wxbf   = (__bf16*)take((size_t)576 * 256 * 2);  // 520 -> pad to 9 strips
  __bf16* Woutbf = (__bf16*)take((size_t)128 * 256 * 2);
  __bf16* cfbf   = (__bf16*)take((size_t)512 * 256 * 2);
  __bf16* cobf   = (__bf16*)take((size_t)128 * 256 * 2);

  float* h1 = xz;   // alias: xz dead once both directions done
  float* h2 = dtv;  // alias: dt dead after scans

  // x -> bf16, FFN weights -> bf16 (once)
  launch_cvt(x, xbf, MTOT * 128, MTOT * 128, stream);
  launch_cvt(convf_w, cfbf, 512 * 256, 512 * 256, stream);
  launch_cvt(convo_w, cobf, 128 * 256, 128 * 256, stream);

  const int NEL = MTOT * 256;
  const int EB  = ceil_div(NEL, 256);

  // ---- two directions, sequential (workspace reuse; stream order = dep order) ----
  for (int dir = 0; dir < 2; ++dir) {
    const int ib = 1 + dir * 9;
    const float* Win   = (const float*)d_in[ib + 0];
    const float* convw = (const float*)d_in[ib + 1];
    const float* convb = (const float*)d_in[ib + 2];
    const float* Wx    = (const float*)d_in[ib + 3];
    const float* Wdt   = (const float*)d_in[ib + 4];
    const float* bdt   = (const float*)d_in[ib + 5];
    const float* Alog  = (const float*)d_in[ib + 6];
    const float* Dp    = (const float*)d_in[ib + 7];
    const float* Wout  = (const float*)d_in[ib + 8];
    const float* scale = dir ? bscale : fscale;

    launch_cvt(Win,  Winbf,  512 * 128, 512 * 128, stream);
    launch_cvt(Wx,   Wxbf,   520 * 256, 576 * 256, stream);  // zero-pad rows 520..575
    launch_cvt(Wout, Woutbf, 128 * 256, 128 * 256, stream);

    // xz = (dir? flip(x) : x) @ Win^T            [8192 x 512]
    launch_gemm(xbf, 128, dir, Winbf, 128, nullptr, xz, 512, MTOT, 512, 128, stream);

    // u = silu(causal_conv4(xi))                 [8192 x 256] f32 + bf16
    conv_silu_kernel<<<EB, 256, 0, stream>>>(xz, 512, convw, convb, uf, ubf);

    // xdbc = u @ Wx^T                            [8192 x 520]
    launch_gemm(ubf, 256, 0, Wxbf, 256, nullptr, xdbc, 528, MTOT, 520, 256, stream);

    // dt = softplus(xdbc[:, :8] @ Wdt^T + bdt)
    dt_kernel<<<EB, 256, 0, stream>>>(xdbc, 528, Wdt, bdt, dtv);

    // selective scan (+ u*D, * silu(z)) -> ygbf
    scan_kernel<<<BB * 32, 256, 0, stream>>>(dtv, uf,
                                             xdbc + 8,   528,   // Bc
                                             xdbc + 264, 528,   // Cc
                                             xz + 256,   512,   // z
                                             Alog, Dp, ygbf);

    // ytmp = yg @ Wout^T                         [8192 x 128]
    launch_gemm(ygbf, 256, 0, Woutbf, 256, nullptr, ytmp, 128, MTOT, 128, 256, stream);

    // residual + scale -> hcat half
    resid_kernel<<<ceil_div(MTOT * 128, 256), 256, 0, stream>>>(
        ytmp, x, scale, dir, dir * 128, hcat);
  }

  // ---- FFN ----
  // h1 = hcat @ convf_w^T + convf_b              [8192 x 512]
  launch_gemm(hcat, 256, 0, cfbf, 256, convf_b, h1, 512, MTOT, 512, 256, stream);

  // 3-tap dwconv + GLU -> glubf                  [8192 x 256]
  dwglu_kernel<<<EB, 256, 0, stream>>>(h1, dw_w, dw_b, glubf);

  // h2 = glu @ convo_w^T + convo_b               [8192 x 128]
  launch_gemm(glubf, 256, 0, cobf, 256, convo_b, h2, 128, MTOT, 128, 256, stream);

  // grouped RMS norm -> out
  rms_kernel<<<MTOT, 128, 0, stream>>>(h2, gamma, (float*)d_out);
}
